// MultilayerVariationalGRU_13537736917054
// MI455X (gfx1250) — matrix operand order repaired and checked by reading, not compiled, not run on hardware
//
#include <hip/hip_runtime.h>
#include <hip/hip_bf16.h>

typedef __attribute__((ext_vector_type(16))) __bf16 v16bf;
typedef __attribute__((ext_vector_type(8)))  __bf16 v8bf;
typedef __attribute__((ext_vector_type(8)))  float  v8f;

#define B_   128
#define L_   1024
#define C_   256
#define H_   256
#define G3_  768
#define NBLK 48
#define TPB  256
#define WAVES_PER_BLK 8
// LDS weight row stride (bf16 elems): 256 + 8 pad -> 528B rows (16B aligned,
// lane n addresses stride 132 dwords -> distinct banks for n=0..15).
#define WPAD 264

__device__ __forceinline__ float fast_sigmoid(float x) {
  return 1.0f / (1.0f + __expf(-x));
}
__device__ __forceinline__ float fast_tanh(float x) {
  return 1.0f - 2.0f / (__expf(2.0f * x) + 1.0f);
}

// A-operand fragment (16x32 bf16 MxK): lane holds K=k0..k0+15 in elems 0..15
// (k0 pre-shifted +16 for lanes 16..31).
__device__ __forceinline__ v16bf load_a_frag(const __bf16* p) {
  union { v16bf v; v8bf h[2]; } u;
  u.h[0] = *(const v8bf*)(p);
  u.h[1] = *(const v8bf*)(p + 8);
  return u.v;
}
// B-operand fragment: 16 contiguous bf16 (K-major within W row), 16B aligned.
__device__ __forceinline__ v16bf load_b_frag(const __bf16* p) {
  union { v16bf v; v8bf h[2]; } u;
  u.h[0] = *(const v8bf*)(p);
  u.h[1] = *(const v8bf*)(p + 8);
  return u.v;
}

// Grid-wide generation barrier; counters re-zeroed by prep kernel each call.
__device__ __forceinline__ void grid_barrier(unsigned* cnt, unsigned* gen) {
  __threadfence();
  __syncthreads();
  if (threadIdx.x == 0) {
    unsigned g = __hip_atomic_load(gen, __ATOMIC_RELAXED, __HIP_MEMORY_SCOPE_AGENT);
    unsigned a = __hip_atomic_fetch_add(cnt, 1u, __ATOMIC_ACQ_REL, __HIP_MEMORY_SCOPE_AGENT);
    if (a == (unsigned)(gridDim.x - 1)) {
      __hip_atomic_store(cnt, 0u, __ATOMIC_RELAXED, __HIP_MEMORY_SCOPE_AGENT);
      __hip_atomic_fetch_add(gen, 1u, __ATOMIC_ACQ_REL, __HIP_MEMORY_SCOPE_AGENT);
    } else {
      while (__hip_atomic_load(gen, __ATOMIC_RELAXED, __HIP_MEMORY_SCOPE_AGENT) == g)
        __builtin_amdgcn_s_sleep(1);
    }
  }
  __syncthreads();
  __threadfence();
}

// Phase 1: one 16x16 tile of GI and GH per wave (mtile=wave, ntile=blockIdx).
// B-operands come from the LDS-staged weight slice; A-operands are bf16.
__device__ __forceinline__ void gru_gates_wmma(
    const __bf16* __restrict__ Agi, long sgi, const __bf16* __restrict__ Agh,
    const __bf16* Wih, const __bf16* Whh,   // LDS, row stride WPAD
    float bi, float bh,
    float* Rb, float* Zb, float* INb, float* HNb) {
  const int lane  = threadIdx.x & 31;
  const int mtile = threadIdx.x >> 5;          // wave id = mtile (0..7)
  const int ntile = blockIdx.x;                // 0..47
  const int n     = lane & 15;
  const int rowA  = mtile * 16 + n;
  const int alo   = (lane & 16) ? 16 : 0;      // A K-shift (hi half-wave)
  const int klo   = (lane & 16) ? 16 : 0;      // B K-shift (hi half-wave)

  const __bf16* argi = Agi + (long)rowA * sgi + alo;
  const __bf16* argh = Agh + (long)rowA * H_ + alo;
  const __bf16* wih  = Wih + n * WPAD + klo;
  const __bf16* whh  = Whh + n * WPAD + klo;

  v8f accI = {};
  v8f accH = {};
#pragma unroll
  for (int kc = 0; kc < C_; kc += 32) {
    v16bf a  = load_a_frag(argi + kc);
    v16bf bI = load_b_frag(wih + kc);
    accI = __builtin_amdgcn_wmma_f32_16x16x32_bf16(
        false, a, false, bI, (short)0, accI, false, false);
    v16bf a2 = load_a_frag(argh + kc);
    v16bf bH = load_b_frag(whh + kc);
    accH = __builtin_amdgcn_wmma_f32_16x16x32_bf16(
        false, a2, false, bH, (short)0, accH, false, false);
  }

  const int region = ntile >> 4;               // 0=r 1=z 2=n (wave-uniform)
  const int j      = (ntile & 15) * 16 + n;    // 0..255 within gate
  const int mbase  = mtile * 16 + ((lane & 16) ? 8 : 0);
#pragma unroll
  for (int g = 0; g < 8; ++g) {
    float gi = accI[g] + bi;
    float gh = accH[g] + bh;
    int   idx = (mbase + g) * H_ + j;
    if (region == 0)      Rb[idx] = fast_sigmoid(gi + gh);
    else if (region == 1) Zb[idx] = fast_sigmoid(gi + gh);
    else                { INb[idx] = gi; HNb[idx] = gh; }
  }
}

// Phase 2: elementwise GRU update; maintains f32 state + bf16 shadow.
__device__ __forceinline__ void gru_update(const float* Rb, const float* Zb,
                                           const float* INb, const float* HNb,
                                           float* H, __bf16* Hb,
                                           float* out, int t) {
  const int stride = gridDim.x * blockDim.x;
  for (int i = blockIdx.x * blockDim.x + threadIdx.x; i < B_ * H_; i += stride) {
    float r = Rb[i], z = Zb[i];
    float nn = fast_tanh(INb[i] + r * HNb[i]);
    float hnew = (1.0f - z) * nn + z * H[i];
    H[i]  = hnew;
    Hb[i] = (__bf16)hnew;
    if (out) {
      int b = i >> 8, j = i & 255;
      out[((long)b * L_ + t) * H_ + j] = hnew;
    }
  }
}

__global__ void gru_prep(const float* x, const float* h0in,
                         const float* wih0, const float* whh0,
                         const float* wih1, const float* whh1,
                         float* H0, float* H1, __bf16* Hb0, __bf16* Hb1,
                         __bf16* W0i, __bf16* W0h, __bf16* W1i, __bf16* W1h,
                         __bf16* xb, unsigned* bar) {
  int tid = blockIdx.x * blockDim.x + threadIdx.x;
  int stride = gridDim.x * blockDim.x;
  if (tid == 0) { bar[0] = 0u; bar[1] = 0u; }
  for (int i = tid; i < B_ * H_; i += stride) {
    float a = h0in[i], b = h0in[B_ * H_ + i];
    H0[i] = a;  Hb0[i] = (__bf16)a;
    H1[i] = b;  Hb1[i] = (__bf16)b;
  }
  for (int i = tid; i < G3_ * C_; i += stride) {
    W0i[i] = (__bf16)wih0[i];
    W0h[i] = (__bf16)whh0[i];
    W1i[i] = (__bf16)wih1[i];
    W1h[i] = (__bf16)whh1[i];
  }
  for (long i = tid; i < (long)B_ * L_ * C_; i += stride)
    xb[i] = (__bf16)x[i];
}

__global__ __launch_bounds__(TPB, 1) void gru_persistent(
    const __bf16* __restrict__ xb,
    const float* bih0, const float* bhh0,
    const float* bih1, const float* bhh1,
    const __bf16* W0i, const __bf16* W0h,
    const __bf16* W1i, const __bf16* W1h,
    float* H0, float* H1, __bf16* Hb0, __bf16* Hb1,
    float* Rb, float* Zb, float* INb, float* HNb,
    float* out, unsigned* bar) {
  unsigned* cnt = bar;
  unsigned* gen = bar + 1;

  // Stage this block's 16-column slice of all four weight matrices into LDS.
  // 4 mats x 16 rows x 256 k (bf16), rows padded to WPAD for bank spread.
  __shared__ __bf16 ldsw[4][16][WPAD];
  {
    const int ncol0 = blockIdx.x * 16;
    for (int idx = threadIdx.x; idx < 4 * 16 * 32; idx += TPB) {
      int mat = idx >> 9;          // 0..3
      int rem = idx & 511;
      int row = rem >> 5;          // 0..15
      int kch = rem & 31;          // 16B chunk (8 bf16)
      const __bf16* src = (mat == 0) ? W0i : (mat == 1) ? W0h
                        : (mat == 2) ? W1i : W1h;
      src += (long)(ncol0 + row) * C_ + kch * 8;
      *(int4*)&ldsw[mat][row][kch * 8] = *(const int4*)src;
    }
  }
  __syncthreads();

  // Hoist wave-invariant bias scalars out of the time loop.
  const int col = blockIdx.x * 16 + (threadIdx.x & 15);
  const float bi0 = bih0[col], bh0 = bhh0[col];
  const float bi1 = bih1[col], bh1 = bhh1[col];

  for (int t = 0; t < L_; ++t) {
    // Prefetch next timestep's x slab (bf16, 128 rows x 512B).
    if (t + 1 < L_) {
      int gtid = blockIdx.x * TPB + threadIdx.x;   // 0..12287
      int prow = gtid >> 5;
      if (prow < B_) {
        int seg = (gtid & 31) * 8;
        __builtin_prefetch(xb + (long)prow * ((long)L_ * C_) + (long)(t + 1) * C_ + seg, 0, 0);
      }
    }

    // Layer 0
    gru_gates_wmma(xb + (long)t * C_, (long)L_ * C_, Hb0,
                   &ldsw[0][0][0], &ldsw[1][0][0],
                   bi0, bh0, Rb, Zb, INb, HNb);
    grid_barrier(cnt, gen);
    gru_update(Rb, Zb, INb, HNb, H0, Hb0, nullptr, t);
    grid_barrier(cnt, gen);

    // Layer 1
    gru_gates_wmma(Hb0, (long)H_, Hb1,
                   &ldsw[2][0][0], &ldsw[3][0][0],
                   bi1, bh1, Rb, Zb, INb, HNb);
    grid_barrier(cnt, gen);
    gru_update(Rb, Zb, INb, HNb, H1, Hb1, out, t);
    grid_barrier(cnt, gen);
  }

  // Final hidden states appended after the [B,L,H] output stream.
  float* hf = out + (long)B_ * L_ * H_;
  const int stride = gridDim.x * blockDim.x;
  for (int i = blockIdx.x * blockDim.x + threadIdx.x; i < B_ * H_; i += stride) {
    hf[i] = H0[i];
    hf[B_ * H_ + i] = H1[i];
  }
}

extern "C" void kernel_launch(void* const* d_in, const int* in_sizes, int n_in,
                              void* d_out, int out_size, void* d_ws, size_t ws_size,
                              hipStream_t stream) {
  const float* x    = (const float*)d_in[0];
  const float* h0   = (const float*)d_in[1];
  const float* wih0 = (const float*)d_in[2];
  const float* whh0 = (const float*)d_in[3];
  const float* bih0 = (const float*)d_in[4];
  const float* bhh0 = (const float*)d_in[5];
  const float* wih1 = (const float*)d_in[6];
  const float* whh1 = (const float*)d_in[7];
  const float* bih1 = (const float*)d_in[8];
  const float* bhh1 = (const float*)d_in[9];
  float* out = (float*)d_out;

  // Workspace carving (256B-aligned blocks): ~70 MB total (xb dominates).
  char* p = (char*)d_ws;
  unsigned* bar = (unsigned*)p;            p += 256;
  float* H0  = (float*)p;                  p += (size_t)B_ * H_ * sizeof(float);
  float* H1  = (float*)p;                  p += (size_t)B_ * H_ * sizeof(float);
  float* Rb  = (float*)p;                  p += (size_t)B_ * H_ * sizeof(float);
  float* Zb  = (float*)p;                  p += (size_t)B_ * H_ * sizeof(float);
  float* INb = (float*)p;                  p += (size_t)B_ * H_ * sizeof(float);
  float* HNb = (float*)p;                  p += (size_t)B_ * H_ * sizeof(float);
  __bf16* Hb0 = (__bf16*)p;                p += (size_t)B_ * H_ * sizeof(__bf16);
  __bf16* Hb1 = (__bf16*)p;                p += (size_t)B_ * H_ * sizeof(__bf16);
  __bf16* W0i = (__bf16*)p;                p += (size_t)G3_ * C_ * sizeof(__bf16);
  __bf16* W0h = (__bf16*)p;                p += (size_t)G3_ * C_ * sizeof(__bf16);
  __bf16* W1i = (__bf16*)p;                p += (size_t)G3_ * C_ * sizeof(__bf16);
  __bf16* W1h = (__bf16*)p;                p += (size_t)G3_ * C_ * sizeof(__bf16);
  __bf16* xb  = (__bf16*)p;                p += (size_t)B_ * L_ * C_ * sizeof(__bf16);

  gru_prep<<<2048, TPB, 0, stream>>>(x, h0, wih0, whh0, wih1, whh1,
                                     H0, H1, Hb0, Hb1,
                                     W0i, W0h, W1i, W1h, xb, bar);

  gru_persistent<<<NBLK, TPB, 0, stream>>>(xb, bih0, bhh0, bih1, bhh1,
                                           W0i, W0h, W1i, W1h,
                                           H0, H1, Hb0, Hb1,
                                           Rb, Zb, INb, HNb,
                                           out, bar);
}
